// LSTMNet_63282048139388
// MI455X (gfx1250) — matrix operand order repaired and checked by reading, not compiled, not run on hardware
//
#include <hip/hip_runtime.h>
#include <hip/hip_bf16.h>
#include <math.h>

typedef __bf16 bf16_t;
typedef __attribute__((ext_vector_type(16))) __bf16 v16bf;
typedef __attribute__((ext_vector_type(8)))  __bf16 v8bf;
typedef __attribute__((ext_vector_type(8)))  float  v8f;

#define B_  16
#define T_  1024
#define I_  256
#define H_  512
#define NG_ 2048   // 4*H
#define O_  256
#define NWG_ 32    // persistent workgroups (one per hidden tile of 16)

// ---------------------------------------------------------------------------
// Prep kernels: fp32 -> bf16 conversions / transposed fused weight builds
// ---------------------------------------------------------------------------

// x[B][T][I] (f32) -> xb[T][B][I] (bf16)
__global__ void convert_x_kernel(const float* __restrict__ x,
                                 bf16_t* __restrict__ xb, int n) {
    int idx = blockIdx.x * blockDim.x + threadIdx.x;
    if (idx >= n) return;
    int i = idx % I_;
    int r = idx / I_;
    int b = r % B_;
    int t = r / B_;
    xb[idx] = (bf16_t)x[((size_t)b * T_ + t) * I_ + i];
}

// W0T[n][k] (bf16, 2048x768): k<256 -> Wk0[k][n], else Wr0[k-256][n]
__global__ void build_w0_kernel(const float* __restrict__ Wk0,
                                const float* __restrict__ Wr0,
                                bf16_t* __restrict__ W0T, int n) {
    int idx = blockIdx.x * blockDim.x + threadIdx.x;
    if (idx >= n) return;
    int k  = idx % 768;
    int nr = idx / 768;
    float v = (k < 256) ? Wk0[(size_t)k * NG_ + nr]
                        : Wr0[(size_t)(k - 256) * NG_ + nr];
    W0T[idx] = (bf16_t)v;
}

// W1T[n][k] (bf16, 2048x1024): k<512 -> Wk1[k][n], else Wr1[k-512][n]
__global__ void build_w1_kernel(const float* __restrict__ Wk1,
                                const float* __restrict__ Wr1,
                                bf16_t* __restrict__ W1T, int n) {
    int idx = blockIdx.x * blockDim.x + threadIdx.x;
    if (idx >= n) return;
    int k  = idx % 1024;
    int nr = idx / 1024;
    float v = (k < 512) ? Wk1[(size_t)k * NG_ + nr]
                        : Wr1[(size_t)(k - 512) * NG_ + nr];
    W1T[idx] = (bf16_t)v;
}

// WoT[n][k] (bf16, 256x512) = Wo[k][n]
__global__ void build_wo_kernel(const float* __restrict__ Wo,
                                bf16_t* __restrict__ WoT, int n) {
    int idx = blockIdx.x * blockDim.x + threadIdx.x;
    if (idx >= n) return;
    int k  = idx % H_;
    int nr = idx / H_;
    WoT[idx] = (bf16_t)Wo[(size_t)k * O_ + nr];
}

// zero ping-pong h state (2 x 16x1024 bf16) and the grid-barrier counter
__global__ void init_state_kernel(bf16_t* __restrict__ hcat,
                                  unsigned* __restrict__ bar) {
    int idx = blockIdx.x * blockDim.x + threadIdx.x;
    if (idx < 2 * B_ * 1024) hcat[idx] = (bf16_t)0.f;
    if (idx == 0) *bar = 0u;
}

// ---------------------------------------------------------------------------
// Helpers
// ---------------------------------------------------------------------------
__device__ __forceinline__ float sigmoid_f(float x) {
    return 1.f / (1.f + __expf(-x));
}
__device__ __forceinline__ float tanh_f(float x) {
    return 1.f - 2.f / (1.f + __expf(2.f * x));     // exact identity, v_exp path
}

// Monotonic agent-scope grid barrier (32 workgroups, thread 0 arbitrates)
__device__ __forceinline__ void grid_barrier(unsigned* bar, unsigned target) {
    __threadfence();                 // make this WG's global writes visible
    __syncthreads();
    if (threadIdx.x == 0) {
        __hip_atomic_fetch_add(bar, 1u, __ATOMIC_RELEASE, __HIP_MEMORY_SCOPE_AGENT);
        while (__hip_atomic_load(bar, __ATOMIC_ACQUIRE, __HIP_MEMORY_SCOPE_AGENT) < target)
            __builtin_amdgcn_s_sleep(2);
    }
    __syncthreads();
    __threadfence();                 // acquire: see other WGs' writes
}

// One gate's 16x16 z-tile: acc = bias + [Alo|Ahi](16 x KTOT) @ WT-row-slice.
// KLO/KTOT/strides are compile-time so the lo/hi source select folds away and
// every load becomes global_load_b128 with an immediate offset.
template<int KLO, int KTOT, int LDLO, int LDHI>
__device__ __forceinline__ v8f gate_gemm(const bf16_t* __restrict__ Alo,
                                         const bf16_t* __restrict__ Ahi,
                                         const bf16_t* __restrict__ wrow,
                                         float bval, int am, int akh) {
    v8f acc = {bval, bval, bval, bval, bval, bval, bval, bval};
    const bf16_t* rlo = Alo + (size_t)am * LDLO + akh;
    const bf16_t* rhi = Ahi + (size_t)am * LDHI + akh;
#pragma unroll
    for (int kb = 0; kb < KTOT; kb += 32) {
        const bf16_t* ar = (kb < KLO) ? (rlo + kb) : (rhi + (kb - KLO));
        v8bf a0 = *reinterpret_cast<const v8bf*>(ar);
        v8bf a1 = *reinterpret_cast<const v8bf*>(ar + 16);
        v16bf af = __builtin_shufflevector(a0, a1,
            0,1,2,3,4,5,6,7,8,9,10,11,12,13,14,15);

        v8bf w0 = *reinterpret_cast<const v8bf*>(wrow + kb);
        v8bf w1 = *reinterpret_cast<const v8bf*>(wrow + kb + 16);
        v16bf wf = __builtin_shufflevector(w0, w1,
            0,1,2,3,4,5,6,7,8,9,10,11,12,13,14,15);

        acc = __builtin_amdgcn_wmma_f32_16x16x32_bf16(
            false, af, false, wf, (short)0, acc, false, false);
    }
    return acc;
}

// ---------------------------------------------------------------------------
// Persistent 2-layer LSTM kernel: 32 WGs x 128 threads (4 waves = 4 gates).
// Cell state c lives in registers for the whole sequence; h ping-pongs through
// a small global buffer; layer/step ordering via atomic grid barrier.
// ---------------------------------------------------------------------------
__global__ __launch_bounds__(128)
void lstm_persistent_kernel(const bf16_t* __restrict__ xb,     // [T][B][I]
                            const bf16_t* __restrict__ W0T,    // [NG][768]
                            const bf16_t* __restrict__ W1T,    // [NG][1024]
                            const float*  __restrict__ b0,
                            const float*  __restrict__ b1,
                            bf16_t* __restrict__ hcat0,        // [B][1024]
                            bf16_t* __restrict__ hcat1,        // [B][1024]
                            bf16_t* __restrict__ seq,          // [T][B][H]
                            float* __restrict__ h0f, float* __restrict__ c0f,
                            float* __restrict__ h1f, float* __restrict__ c1f,
                            unsigned* __restrict__ bar)
{
    __shared__ float zs[4][16][16];

    const int lane = threadIdx.x & 31;
    const int wave = threadIdx.x >> 5;        // gate id: 0=i 1=f 2=g 3=o
    const int ht   = blockIdx.x;              // hidden tile 0..31
    const int n0   = wave * H_ + ht * 16;     // z column base for this gate
    const int nl   = lane & 15;
    const int hi16 = lane >> 4;
    const int am   = nl;                      // A fragment M row
    const int akh  = hi16 * 8;                // A fragment K sub-offset

    const bf16_t* w0row = W0T + (size_t)(n0 + nl) * 768  + hi16 * 8;
    const bf16_t* w1row = W1T + (size_t)(n0 + nl) * 1024 + hi16 * 8;
    const float b0v = b0[n0 + nl];
    const float b1v = b1[n0 + nl];

    // this thread's two (m, u) elements of the block's 16x16 tile
    const int e0m = threadIdx.x >> 4;         // m = 0..7
    const int e1m = e0m + 8;                  // m = 8..15
    const int eu  = threadIdx.x & 15;
    const int ug  = ht * 16 + eu;             // global hidden unit
    float c0a = 0.f, c0b = 0.f, c1a = 0.f, c1b = 0.f;

    unsigned phase = 0;

    for (int t = 0; t < T_; ++t) {
        bf16_t* cur = (t & 1) ? hcat1 : hcat0;   // h0(t-1) | h1(t-1)
        bf16_t* nxt = (t & 1) ? hcat0 : hcat1;

        // ===== layer 0: z = [x_t (K 0..255) | h0_prev (K 256..767)] =====
        {
            v8f acc = gate_gemm<256, 768, I_, 1024>(
                xb + (size_t)t * B_ * I_, cur, w0row, b0v, am, akh);
#pragma unroll
            for (int v = 0; v < 8; ++v)
                zs[wave][v + hi16 * 8][nl] = acc[v];
        }
        __syncthreads();
        {
            float zi = zs[0][e0m][eu], zf = zs[1][e0m][eu];
            float zg = zs[2][e0m][eu], zo = zs[3][e0m][eu];
            float cn = sigmoid_f(zf) * c0a + sigmoid_f(zi) * tanh_f(zg);
            float hn = sigmoid_f(zo) * tanh_f(cn);
            c0a = cn;
            nxt[e0m * 1024 + ug] = (bf16_t)hn;
            if (t == T_ - 1) h0f[e0m * H_ + ug] = hn;

            zi = zs[0][e1m][eu]; zf = zs[1][e1m][eu];
            zg = zs[2][e1m][eu]; zo = zs[3][e1m][eu];
            cn = sigmoid_f(zf) * c0b + sigmoid_f(zi) * tanh_f(zg);
            hn = sigmoid_f(zo) * tanh_f(cn);
            c0b = cn;
            nxt[e1m * 1024 + ug] = (bf16_t)hn;
            if (t == T_ - 1) h0f[e1m * H_ + ug] = hn;
        }
        grid_barrier(bar, NWG_ * (++phase));   // h0(t) visible everywhere

        // ===== layer 1: z = [h0_new (K 0..511) | h1_prev (K 512..1023)] =====
        {
            v8f acc = gate_gemm<512, 1024, 1024, 1024>(
                nxt, cur + 512, w1row, b1v, am, akh);
#pragma unroll
            for (int v = 0; v < 8; ++v)
                zs[wave][v + hi16 * 8][nl] = acc[v];
        }
        __syncthreads();
        {
            bf16_t* sq = seq + (size_t)t * B_ * H_;

            float zi = zs[0][e0m][eu], zf = zs[1][e0m][eu];
            float zg = zs[2][e0m][eu], zo = zs[3][e0m][eu];
            float cn = sigmoid_f(zf) * c1a + sigmoid_f(zi) * tanh_f(zg);
            float hn = sigmoid_f(zo) * tanh_f(cn);
            c1a = cn;
            bf16_t hb = (bf16_t)hn;
            nxt[e0m * 1024 + 512 + ug] = hb;
            sq[e0m * H_ + ug] = hb;
            if (t == T_ - 1) h1f[e0m * H_ + ug] = hn;

            zi = zs[0][e1m][eu]; zf = zs[1][e1m][eu];
            zg = zs[2][e1m][eu]; zo = zs[3][e1m][eu];
            cn = sigmoid_f(zf) * c1b + sigmoid_f(zi) * tanh_f(zg);
            hn = sigmoid_f(zo) * tanh_f(cn);
            c1b = cn;
            hb = (bf16_t)hn;
            nxt[e1m * 1024 + 512 + ug] = hb;
            sq[e1m * H_ + ug] = hb;
            if (t == T_ - 1) h1f[e1m * H_ + ug] = hn;
        }
        grid_barrier(bar, NWG_ * (++phase));   // h1(t) visible everywhere
    }

    // final cell states (kept in registers for the whole sequence)
    c0f[e0m * H_ + ug] = c0a;
    c0f[e1m * H_ + ug] = c0b;
    c1f[e0m * H_ + ug] = c1a;
    c1f[e1m * H_ + ug] = c1b;
}

// ---------------------------------------------------------------------------
// Output projection: out[b][t][:] = seq[t][b][:] @ Wo + bo
// ---------------------------------------------------------------------------
__global__ __launch_bounds__(128)
void proj_kernel(const bf16_t* __restrict__ seq,   // [T][B][H] bf16
                 const bf16_t* __restrict__ WoT,   // [O][H] bf16
                 const float*  __restrict__ bo,    // [O]
                 float* __restrict__ out)          // [B][T][O]
{
    const int lane = threadIdx.x & 31;
    const int wave = threadIdx.x >> 5;
    const int tile = blockIdx.x * 4 + wave;   // 0 .. T*16-1
    const int t  = tile >> 4;
    const int nt = tile & 15;

    const int nl   = lane & 15;
    const int hi16 = lane >> 4;

    const float bval = bo[nt * 16 + nl];
    v8f acc = {bval, bval, bval, bval, bval, bval, bval, bval};

    const bf16_t* arow = seq + (size_t)t * (B_ * H_) + (size_t)nl * H_ + hi16 * 8;
    const bf16_t* brow = WoT + (size_t)(nt * 16 + nl) * H_ + hi16 * 8;

#pragma unroll
    for (int kb = 0; kb < H_; kb += 32) {
        v8bf a0 = *reinterpret_cast<const v8bf*>(arow + kb);
        v8bf a1 = *reinterpret_cast<const v8bf*>(arow + kb + 16);
        v16bf af = __builtin_shufflevector(a0, a1,
            0,1,2,3,4,5,6,7,8,9,10,11,12,13,14,15);

        v8bf w0 = *reinterpret_cast<const v8bf*>(brow + kb);
        v8bf w1 = *reinterpret_cast<const v8bf*>(brow + kb + 16);
        v16bf wf = __builtin_shufflevector(w0, w1,
            0,1,2,3,4,5,6,7,8,9,10,11,12,13,14,15);

        acc = __builtin_amdgcn_wmma_f32_16x16x32_bf16(
            false, af, false, wf, (short)0, acc, false, false);
    }

#pragma unroll
    for (int v = 0; v < 8; ++v) {
        int m = v + hi16 * 8;     // batch row
        out[((size_t)m * T_ + t) * O_ + nt * 16 + nl] = acc[v];
    }
}

// ---------------------------------------------------------------------------
extern "C" void kernel_launch(void* const* d_in, const int* in_sizes, int n_in,
                              void* d_out, int out_size, void* d_ws, size_t ws_size,
                              hipStream_t stream) {
    (void)in_sizes; (void)n_in; (void)out_size; (void)ws_size;

    const float* x   = (const float*)d_in[0];
    const float* Wk0 = (const float*)d_in[1];
    const float* Wr0 = (const float*)d_in[2];
    const float* b0  = (const float*)d_in[3];
    const float* Wk1 = (const float*)d_in[4];
    const float* Wr1 = (const float*)d_in[5];
    const float* b1  = (const float*)d_in[6];
    const float* Wo  = (const float*)d_in[7];
    const float* bo  = (const float*)d_in[8];

    char* ws = (char*)d_ws;
    size_t off = 0;
    auto take = [&](size_t bytes) {
        char* p = ws + off;
        off = (off + bytes + 255) & ~(size_t)255;
        return p;
    };
    bf16_t*   xb    = (bf16_t*)take((size_t)T_ * B_ * I_ * 2);   //  8 MB
    bf16_t*   W0T   = (bf16_t*)take((size_t)NG_ * 768  * 2);     //  3 MB
    bf16_t*   W1T   = (bf16_t*)take((size_t)NG_ * 1024 * 2);     //  4 MB
    bf16_t*   WoT   = (bf16_t*)take((size_t)O_ * H_ * 2);        // 256 KB
    bf16_t*   hcat0 = (bf16_t*)take((size_t)B_ * 1024 * 2);      // ping
    bf16_t*   hcat1 = (bf16_t*)take((size_t)B_ * 1024 * 2);      // pong
    bf16_t*   seq   = (bf16_t*)take((size_t)T_ * B_ * H_ * 2);   // 16 MB
    unsigned* bar   = (unsigned*)take(256);

    float* out = (float*)d_out;                  // [B][T][O]
    float* h0f = out + (size_t)B_ * T_ * O_;     // h_0 [B][H]
    float* c0f = h0f + B_ * H_;                  // c_0
    float* h1f = c0f + B_ * H_;                  // h_1
    float* c1f = h1f + B_ * H_;                  // c_1

    // ---- prep (inside graph, deterministic) ----
    { int n = T_ * B_ * I_; convert_x_kernel<<<(n + 255) / 256, 256, 0, stream>>>(x, xb, n); }
    { int n = NG_ * 768;    build_w0_kernel<<<(n + 255) / 256, 256, 0, stream>>>(Wk0, Wr0, W0T, n); }
    { int n = NG_ * 1024;   build_w1_kernel<<<(n + 255) / 256, 256, 0, stream>>>(Wk1, Wr1, W1T, n); }
    { int n = O_ * H_;      build_wo_kernel<<<(n + 255) / 256, 256, 0, stream>>>(Wo, WoT, n); }
    init_state_kernel<<<(2 * B_ * 1024 + 255) / 256, 256, 0, stream>>>(hcat0, bar);

    // ---- whole recurrence in one persistent kernel ----
    lstm_persistent_kernel<<<NWG_, 128, 0, stream>>>(
        xb, W0T, W1T, b0, b1, hcat0, hcat1, seq,
        h0f, c0f, h1f, c1f, bar);

    // ---- output projection over whole sequence ----
    proj_kernel<<<(T_ * 16) / 4, 128, 0, stream>>>(seq, WoT, bo, out);
}
